// AsymmetricJointBlock_58703613002348
// MI455X (gfx1250) — matrix operand order repaired and checked by reading, not compiled, not run on hardware
//
#include <hip/hip_runtime.h>

// ============================================================================
// MI455X (gfx1250, wave32) implementation of the asymmetric joint DiT block.
//
// Strategy (from roofline): ~830MB of fp32 weights each used exactly once ->
// the block is weight-bandwidth bound (~36us at 23.3 TB/s). So: stream fp32
// weights once, convert to bf16 in the global->LDS staging path (free), and
// run all matmuls on v_wmma_f32_16x16x32_bf16 (16384 FLOP/inst, f32 accum).
// Attention is a flash-style WMMA kernel computing S^T = K*Q^T so the
// softmax is per-lane and the S^T output layout IS the A-fragment layout
// for the P*V WMMA (no transpose of P needed).
// ============================================================================

typedef __attribute__((ext_vector_type(16))) __bf16 v16bf;
typedef __attribute__((ext_vector_type(8)))  float  v8f;

union FragB {
    v16bf v;
    uint4 q[2];
    unsigned short us[16];
};

__device__ __forceinline__ unsigned short f2bf(float f) {
    union { float f; unsigned int u; } v; v.f = f;
    unsigned int u = v.u;
    return (unsigned short)((u + 0x7FFFu + ((u >> 16) & 1u)) >> 16);
}
__device__ __forceinline__ float bf2f(unsigned short h) {
    union { unsigned int u; float f; } v; v.u = ((unsigned int)h) << 16;
    return v.f;
}
__device__ __forceinline__ v8f vzero8() {
    v8f z;
#pragma unroll
    for (int i = 0; i < 8; ++i) z[i] = 0.0f;
    return z;
}
__device__ __forceinline__ v8f wmma_bf16(FragB a, FragB b, v8f c) {
    return __builtin_amdgcn_wmma_f32_16x16x32_bf16(false, a.v, false, b.v,
                                                   (short)0, c, false, false);
}

// ---------------------------------------------------------------------------
// silu(c)
__global__ void silu_kernel(const float* __restrict__ c, float* __restrict__ cs, int n) {
    int i = blockIdx.x * blockDim.x + threadIdx.x;
    if (i < n) { float v = c[i]; cs[i] = v / (1.0f + __expf(-v)); }
}

// mod vectors: one wave per output, dot(c_silu, W_row) + b
__launch_bounds__(256)
__global__ void modvec_kernel(const float* __restrict__ cs,
                              const float* __restrict__ wx, const float* __restrict__ bx,
                              const float* __restrict__ wy, const float* __restrict__ by,
                              float* __restrict__ modx, float* __restrict__ mody, int K) {
    const int g    = blockIdx.x * 8 + (threadIdx.x >> 5);
    const int lane = threadIdx.x & 31;
    const float* wrow; const float* bias; float* outp; int o;
    if (g < 12288) { o = g;          wrow = wx + (size_t)o * K; bias = bx; outp = modx; }
    else           { o = g - 12288;  wrow = wy + (size_t)o * K; bias = by; outp = mody; }
    float s = 0.0f;
    for (int k = lane; k < K; k += 32) s += wrow[k] * cs[k];
#pragma unroll
    for (int off = 16; off; off >>= 1) s += __shfl_xor(s, off, 32);
    if (lane == 0) outp[o] = s + bias[o];
}

// modulated rmsnorm -> bf16:  out = x * rsqrt(mean(x^2)+eps) * (1+scale[col])
__launch_bounds__(256)
__global__ void modnorm_kernel(const float* __restrict__ x, const float* __restrict__ scale,
                               unsigned short* __restrict__ out, int cols) {
    const int row = blockIdx.x;
    const float* xr = x + (size_t)row * cols;
    __shared__ float red[8];
    float s = 0.0f;
    for (int c = threadIdx.x; c < cols; c += 256) { float v = xr[c]; s += v * v; }
#pragma unroll
    for (int off = 16; off; off >>= 1) s += __shfl_xor(s, off, 32);
    if ((threadIdx.x & 31) == 0) red[threadIdx.x >> 5] = s;
    __syncthreads();
    if (threadIdx.x < 8) {
        float t = red[threadIdx.x];
#pragma unroll
        for (int off = 4; off; off >>= 1) t += __shfl_xor(t, off, 32);
        if (threadIdx.x == 0) red[0] = t;
    }
    __syncthreads();
    const float inv = rsqrtf(red[0] / (float)cols + 1e-6f);
    unsigned short* orow = out + (size_t)row * cols;
    for (int c = threadIdx.x; c < cols; c += 256)
        orow[c] = f2bf(xr[c] * inv * (1.0f + scale[c]));
}

// residual tanh gated rmsnorm: out = x + r * rsqrt(mean(r^2)+eps) * tanh(gate[col])
__launch_bounds__(256)
__global__ void residual_kernel(const float* __restrict__ x, const float* __restrict__ r,
                                const float* __restrict__ gate, float* __restrict__ out, int cols) {
    const int row = blockIdx.x;
    const float* rr = r + (size_t)row * cols;
    const float* xr = x + (size_t)row * cols;
    __shared__ float red[8];
    float s = 0.0f;
    for (int c = threadIdx.x; c < cols; c += 256) { float v = rr[c]; s += v * v; }
#pragma unroll
    for (int off = 16; off; off >>= 1) s += __shfl_xor(s, off, 32);
    if ((threadIdx.x & 31) == 0) red[threadIdx.x >> 5] = s;
    __syncthreads();
    if (threadIdx.x < 8) {
        float t = red[threadIdx.x];
#pragma unroll
        for (int off = 4; off; off >>= 1) t += __shfl_xor(t, off, 32);
        if (threadIdx.x == 0) red[0] = t;
    }
    __syncthreads();
    const float inv = rsqrtf(red[0] / (float)cols + 1e-6f);
    float* orow = out + (size_t)row * cols;
    for (int c = threadIdx.x; c < cols; c += 256)
        orow[c] = xr[c] + rr[c] * inv * tanhf(gate[c]);
}

// silu(h)*gate on bf16 hidden buffer [rows, 2F] -> [rows, F]
__global__ void silugate_kernel(const unsigned short* __restrict__ hg,
                                unsigned short* __restrict__ act, int rows, int F) {
    const size_t n = (size_t)rows * F;
    for (size_t i = (size_t)blockIdx.x * blockDim.x + threadIdx.x; i < n;
         i += (size_t)gridDim.x * blockDim.x) {
        size_t row = i / F, col = i - row * F;
        float hv = bf2f(hg[row * (size_t)(2 * F) + col]);
        float gv = bf2f(hg[row * (size_t)(2 * F) + F + col]);
        act[i] = f2bf((hv / (1.0f + __expf(-hv))) * gv);
    }
}

// ---------------------------------------------------------------------------
// GEMM: out[M,N] = A_bf16[M,K] @ W_f32[N,K]^T (+bias). W converted to bf16 in
// the staging path. CTA tile 128x128, K-step 32, 8 waves each doing 4x2
// 16x16 WMMA tiles. blockIdx.x = M tile (co-resident CTAs share W via L2).
#define LDS_STRIDE 40
__launch_bounds__(256)
__global__ void gemm_bf16(const unsigned short* __restrict__ A,
                          const float* __restrict__ W,
                          const float* __restrict__ bias,
                          void* __restrict__ out,
                          int M, int N, int K, int out_bf16) {
    __shared__ __align__(16) unsigned short sA[128 * LDS_STRIDE];
    __shared__ __align__(16) unsigned short sB[128 * LDS_STRIDE];
    const int tid  = threadIdx.x;
    const int lane = tid & 31;
    const int wave = tid >> 5;      // 0..7
    const int wm   = wave >> 2;     // 0..1
    const int wn   = wave & 3;      // 0..3
    const int idx  = lane & 15;
    const int lhf  = lane >> 4;
    const int tileM = blockIdx.x * 128;
    const int tileN = blockIdx.y * 128;
    const int sr = tid >> 1;            // staging row 0..127
    const int sc = (tid & 1) * 16;      // staging col 0/16

    v8f acc[4][2];
#pragma unroll
    for (int i = 0; i < 4; ++i)
#pragma unroll
        for (int j = 0; j < 2; ++j) acc[i][j] = vzero8();

    for (int k0 = 0; k0 < K; k0 += 32) {
        {   // stage A (bf16 copy)
            const unsigned short* src = A + (size_t)(tileM + sr) * K + k0 + sc;
            *(uint4*)(&sA[sr * LDS_STRIDE + sc])     = *(const uint4*)(src);
            *(uint4*)(&sA[sr * LDS_STRIDE + sc + 8]) = *(const uint4*)(src + 8);
        }
        {   // stage W (f32 -> bf16)
            const float* src = W + (size_t)(tileN + sr) * K + k0 + sc;
            if (k0 + 32 < K) __builtin_prefetch(src + 32, 0, 1);
            union { unsigned short us[16]; uint4 q[2]; } tmp;
            const float4* s4 = (const float4*)src;
#pragma unroll
            for (int j = 0; j < 4; ++j) {
                float4 f = s4[j];
                tmp.us[4 * j + 0] = f2bf(f.x);
                tmp.us[4 * j + 1] = f2bf(f.y);
                tmp.us[4 * j + 2] = f2bf(f.z);
                tmp.us[4 * j + 3] = f2bf(f.w);
            }
            *(uint4*)(&sB[sr * LDS_STRIDE + sc])     = tmp.q[0];
            *(uint4*)(&sB[sr * LDS_STRIDE + sc + 8]) = tmp.q[1];
        }
        __syncthreads();
        FragB a[4], b[2];
#pragma unroll
        for (int i = 0; i < 4; ++i) {   // A frag: K split {h*8+0..7, 16+h*8+0..7}
            const unsigned short* p = &sA[(wm * 64 + i * 16 + idx) * LDS_STRIDE + lhf * 8];
            a[i].q[0] = *(const uint4*)(p);
            a[i].q[1] = *(const uint4*)(p + 16);
        }
#pragma unroll
        for (int j = 0; j < 2; ++j) {   // B frag: K split {h*8+0..7, 16+h*8+0..7}
            const unsigned short* p = &sB[(wn * 32 + j * 16 + idx) * LDS_STRIDE + lhf * 8];
            b[j].q[0] = *(const uint4*)(p);
            b[j].q[1] = *(const uint4*)(p + 16);
        }
#pragma unroll
        for (int i = 0; i < 4; ++i)
#pragma unroll
            for (int j = 0; j < 2; ++j) acc[i][j] = wmma_bf16(a[i], b[j], acc[i][j]);
        __syncthreads();
    }

#pragma unroll
    for (int i = 0; i < 4; ++i) {
#pragma unroll
        for (int j = 0; j < 2; ++j) {
            const int col = tileN + wn * 32 + j * 16 + idx;
            const float bv = bias ? bias[col] : 0.0f;
#pragma unroll
            for (int r = 0; r < 8; ++r) {
                const int row = tileM + wm * 64 + i * 16 + lhf * 8 + r;
                const float v = acc[i][j][r] + bv;
                if (out_bf16) ((unsigned short*)out)[(size_t)row * N + col] = f2bf(v);
                else          ((float*)out)[(size_t)row * N + col] = v;
            }
        }
    }
}

// ---------------------------------------------------------------------------
// q/k prep: rmsnorm (eps 1e-5, per-head weight), RoPE for x tokens, fold
// 1/sqrt(HD) into q, pack q/k/v as bf16 in [H][S][HD] layout. One wave per (s,h).
__launch_bounds__(256)
__global__ void qkprep_kernel(const float* __restrict__ qkvx, const float* __restrict__ qkvy,
                              const float* __restrict__ qwx, const float* __restrict__ kwx,
                              const float* __restrict__ qwy, const float* __restrict__ kwy,
                              const float* __restrict__ rc, const float* __restrict__ rs,
                              unsigned short* __restrict__ qb, unsigned short* __restrict__ kb,
                              unsigned short* __restrict__ vb) {
    const int g    = blockIdx.x * 8 + (threadIdx.x >> 5);
    const int lane = threadIdx.x & 31;
    const int s = g / 24, h = g % 24;
    const int d0 = lane * 4;
    const float *src, *qw, *kw;
    bool rope;
    if (s < 2048) { src = qkvx + (size_t)s * 9216;        qw = qwx; kw = kwx; rope = true; }
    else          { src = qkvy + (size_t)(s - 2048) * 9216; qw = qwy; kw = kwy; rope = false; }
    float4 q4 = *(const float4*)(src + h * 128 + d0);
    float4 k4 = *(const float4*)(src + 3072 + h * 128 + d0);
    float4 v4 = *(const float4*)(src + 6144 + h * 128 + d0);
    float sq = q4.x*q4.x + q4.y*q4.y + q4.z*q4.z + q4.w*q4.w;
    float sk = k4.x*k4.x + k4.y*k4.y + k4.z*k4.z + k4.w*k4.w;
#pragma unroll
    for (int off = 16; off; off >>= 1) {
        sq += __shfl_xor(sq, off, 32);
        sk += __shfl_xor(sk, off, 32);
    }
    const float iq = rsqrtf(sq * (1.0f / 128.0f) + 1e-5f);
    const float ik = rsqrtf(sk * (1.0f / 128.0f) + 1e-5f);
    float4 wq = *(const float4*)(qw + d0);
    float4 wk = *(const float4*)(kw + d0);
    float qx = q4.x*iq*wq.x, qy = q4.y*iq*wq.y, qz = q4.z*iq*wq.z, qw_ = q4.w*iq*wq.w;
    float kx = k4.x*ik*wk.x, ky = k4.y*ik*wk.y, kz = k4.z*ik*wk.z, kw_ = k4.w*ik*wk.w;
    if (rope) {
        const size_t ro = ((size_t)s * 24 + h) * 64 + lane * 2;
        const float c0 = rc[ro], c1 = rc[ro + 1], s0 = rs[ro], s1 = rs[ro + 1];
        float e, o;
        e = qx; o = qy;  qx = e*c0 - o*s0; qy  = e*s0 + o*c0;
        e = qz; o = qw_; qz = e*c1 - o*s1; qw_ = e*s1 + o*c1;
        e = kx; o = ky;  kx = e*c0 - o*s0; ky  = e*s0 + o*c0;
        e = kz; o = kw_; kz = e*c1 - o*s1; kw_ = e*s1 + o*c1;
    }
    const float scl = 0.08838834764831845f; // 1/sqrt(128)
    const size_t ob = ((size_t)h * 2304 + s) * 128 + d0;
    *(ushort4*)(qb + ob) = make_ushort4(f2bf(qx*scl), f2bf(qy*scl), f2bf(qz*scl), f2bf(qw_*scl));
    *(ushort4*)(kb + ob) = make_ushort4(f2bf(kx), f2bf(ky), f2bf(kz), f2bf(kw_));
    *(ushort4*)(vb + ob) = make_ushort4(f2bf(v4.x), f2bf(v4.y), f2bf(v4.z), f2bf(v4.w));
}

// ---------------------------------------------------------------------------
// Flash attention, all-WMMA. Grid (24 heads, 18 query blocks of 128).
// Each wave owns 16 queries; 32-key tiles of K and V^T staged in LDS.
// S^T = K*Q^T puts softmax per-lane; its D layout equals the P A-fragment.
__launch_bounds__(256)
__global__ void attn_kernel(const unsigned short* __restrict__ qb,
                            const unsigned short* __restrict__ kb,
                            const unsigned short* __restrict__ vb,
                            unsigned short* __restrict__ out) {
    __shared__ __align__(16) unsigned short sK[32 * 136];
    __shared__ __align__(16) unsigned short sVT[128 * 40];
    const int h    = blockIdx.x;
    const int tid  = threadIdx.x;
    const int lane = tid & 31;
    const int wave = tid >> 5;
    const int idx  = lane & 15;
    const int lhf  = lane >> 4;
    const int q0   = blockIdx.y * 128 + wave * 16;
    const size_t headoff = (size_t)h * 2304 * 128;

    FragB qf[4];   // B operand: split-K pattern {h*8+0..7, 16+h*8+0..7}
    {
        const unsigned short* qrow = qb + headoff + (size_t)(q0 + idx) * 128;
#pragma unroll
        for (int c = 0; c < 4; ++c) {
            qf[c].q[0] = *(const uint4*)(qrow + c * 32 + lhf * 8);
            qf[c].q[1] = *(const uint4*)(qrow + c * 32 + lhf * 8 + 16);
        }
    }
    v8f O[8];
#pragma unroll
    for (int t = 0; t < 8; ++t) O[t] = vzero8();
    float m_run = -3.0e38f, l_run = 0.0f;

    const int sr = tid >> 3;          // key row 0..31
    const int sc = (tid & 7) * 16;    // dim seg

    for (int kt = 0; kt < 2304; kt += 32) {
        {   // stage K rows (row-major bf16 copy)
            const unsigned short* krow = kb + headoff + (size_t)(kt + sr) * 128 + sc;
            *(uint4*)(&sK[sr * 136 + sc])     = *(const uint4*)(krow);
            *(uint4*)(&sK[sr * 136 + sc + 8]) = *(const uint4*)(krow + 8);
        }
        {   // stage V transposed: sVT[d][key]
            const unsigned short* vrow = vb + headoff + (size_t)(kt + sr) * 128 + sc;
            union { uint4 q[2]; unsigned short us[16]; } tv;
            tv.q[0] = *(const uint4*)(vrow);
            tv.q[1] = *(const uint4*)(vrow + 8);
#pragma unroll
            for (int j = 0; j < 16; ++j) sVT[(sc + j) * 40 + sr] = tv.us[j];
        }
        __syncthreads();

        // S^T tiles: keys 0-15 and 16-31 (x queries 0-15)
        v8f st[2];
        st[0] = vzero8(); st[1] = vzero8();
#pragma unroll
        for (int stt = 0; stt < 2; ++stt) {
#pragma unroll
            for (int c = 0; c < 4; ++c) {
                FragB kf;   // A operand: split-K pattern
                const unsigned short* p = &sK[(stt * 16 + idx) * 136 + c * 32 + lhf * 8];
                kf.q[0] = *(const uint4*)(p);
                kf.q[1] = *(const uint4*)(p + 16);
                st[stt] = wmma_bf16(kf, qf[c], st[stt]);
            }
        }
        // online softmax (per lane = one query column; pair with lane^16)
        float mloc = -3.0e38f;
#pragma unroll
        for (int r = 0; r < 8; ++r) mloc = fmaxf(mloc, fmaxf(st[0][r], st[1][r]));
        mloc = fmaxf(mloc, __shfl_xor(mloc, 16, 32));
        const float m_new = fmaxf(m_run, mloc);
        const float alpha = __expf(m_run - m_new);
        FragB pf;
        float ls = 0.0f;
#pragma unroll
        for (int r = 0; r < 8; ++r) {
            float p0 = __expf(st[0][r] - m_new);
            float p1 = __expf(st[1][r] - m_new);
            ls += p0 + p1;
            pf.us[r]     = f2bf(p0);   // keys {lhf*8 + r}
            pf.us[8 + r] = f2bf(p1);   // keys {16 + lhf*8 + r}
        }
        ls += __shfl_xor(ls, 16, 32);
        l_run = l_run * alpha + ls;
        m_run = m_new;
        // rescale O (row q = lhf*8 + r; alpha lives in lane q)
#pragma unroll
        for (int r = 0; r < 8; ++r) {
            const float ar = __shfl(alpha, lhf * 8 + r, 32);
#pragma unroll
            for (int t = 0; t < 8; ++t) O[t][r] *= ar;
        }
        // O += P * V  (B operand from sVT rows, split-K keys {h*8+0..7, 16+h*8+0..7})
#pragma unroll
        for (int t = 0; t < 8; ++t) {
            FragB vf;
            const unsigned short* p = &sVT[(t * 16 + idx) * 40 + lhf * 8];
            vf.q[0] = *(const uint4*)(p);
            vf.q[1] = *(const uint4*)(p + 16);
            O[t] = wmma_bf16(pf, vf, O[t]);
        }
        __syncthreads();
    }
    // finalize: divide by l, write bf16 to [s][h*128+d]
#pragma unroll
    for (int r = 0; r < 8; ++r) {
        const float lr  = __shfl(l_run, lhf * 8 + r, 32);
        const float inv = 1.0f / lr;
        const int qrow  = q0 + lhf * 8 + r;
        unsigned short* orow = out + (size_t)qrow * 3072 + h * 128;
#pragma unroll
        for (int t = 0; t < 8; ++t) orow[t * 16 + idx] = f2bf(O[t][r] * inv);
    }
}

// ============================================================================
extern "C" void kernel_launch(void* const* d_in, const int* in_sizes, int n_in,
                              void* d_out, int out_size, void* d_ws, size_t ws_size,
                              hipStream_t stream) {
    const float* x        = (const float*)d_in[0];
    const float* c        = (const float*)d_in[1];
    const float* y        = (const float*)d_in[2];
    const float* rope_cos = (const float*)d_in[3];
    const float* rope_sin = (const float*)d_in[4];
    const float* mod_x_w  = (const float*)d_in[5];
    const float* mod_x_b  = (const float*)d_in[6];
    const float* mod_y_w  = (const float*)d_in[7];
    const float* mod_y_b  = (const float*)d_in[8];
    const float* qkv_x_w  = (const float*)d_in[9];
    const float* qkv_x_b  = (const float*)d_in[10];
    const float* qkv_y_w  = (const float*)d_in[11];
    const float* qkv_y_b  = (const float*)d_in[12];
    const float* q_norm_x_w = (const float*)d_in[13];
    const float* k_norm_x_w = (const float*)d_in[14];
    const float* q_norm_y_w = (const float*)d_in[15];
    const float* k_norm_y_w = (const float*)d_in[16];
    const float* proj_x_w = (const float*)d_in[17];
    const float* proj_x_b = (const float*)d_in[18];
    const float* proj_y_w = (const float*)d_in[19];
    const float* proj_y_b = (const float*)d_in[20];
    const float* mlp_x_w1 = (const float*)d_in[21];
    const float* mlp_x_w2 = (const float*)d_in[22];
    const float* mlp_y_w1 = (const float*)d_in[23];
    const float* mlp_y_w2 = (const float*)d_in[24];

    char* ws = (char*)d_ws;
    const size_t MB = 1ull << 20;
    float* cs   = (float*)(ws + 0);
    float* modx = (float*)(ws + 16384);     // 12288 f32: sx,gx,smx,gmx
    float* mody = (float*)(ws + 81920);     //  6144 f32: sy,gy,smy,gmy
    unsigned short* xm    = (unsigned short*)(ws + 1   * MB); // 2048x3072 bf16 (reused as xm2)
    unsigned short* ym    = (unsigned short*)(ws + 14  * MB); //  256x1536 bf16 (reused as ym2)
    float*          qkvx  = (float*)(ws + 15  * MB);          // 2048x9216 f32
    float*          qkvy  = (float*)(ws + 91  * MB);          //  256x9216 f32
    unsigned short* qb    = (unsigned short*)(ws + 106 * MB); // [24][2304][128] bf16
    unsigned short* kb    = (unsigned short*)(ws + 121 * MB);
    unsigned short* vb    = (unsigned short*)(ws + 136 * MB);
    unsigned short* attnb = (unsigned short*)(ws + 151 * MB); // 2304x3072 bf16
    unsigned short* hgx   = (unsigned short*)(ws + 166 * MB); // 2048x16384 bf16 (67MB)
    // region reuse (producers dead before reuse):
    float*          xattn = (float*)(ws + 15 * MB);           // over qkvx
    unsigned short* actx  = (unsigned short*)(ws + 47 * MB);  // 2048x8192 bf16
    float*          mlpx  = (float*)(ws + 15 * MB);           // over xattn
    float*          yattn = (float*)(ws + 91 * MB);           // over qkvy
    unsigned short* hgy   = (unsigned short*)(ws + 93 * MB);  // 256x8192 bf16
    unsigned short* acty  = (unsigned short*)(ws + 98 * MB);  // 256x4096 bf16
    float*          mlpy  = (float*)(ws + 101 * MB);

    float* outx = (float*)d_out;                    // 2048x3072
    float* outy = outx + (size_t)2048 * 3072;       //  256x1536

    // 1. modulation vectors
    silu_kernel<<<12, 256, 0, stream>>>(c, cs, 3072);
    modvec_kernel<<<2304, 256, 0, stream>>>(cs, mod_x_w, mod_x_b, mod_y_w, mod_y_b,
                                            modx, mody, 3072);
    // 2. modulated rmsnorm -> bf16 activations
    modnorm_kernel<<<2048, 256, 0, stream>>>(x, modx, xm, 3072);
    modnorm_kernel<<<256,  256, 0, stream>>>(y, mody, ym, 1536);
    // 3. QKV GEMMs
    gemm_bf16<<<dim3(16, 72), 256, 0, stream>>>(xm, qkv_x_w, qkv_x_b, qkvx, 2048, 9216, 3072, 0);
    gemm_bf16<<<dim3(2,  72), 256, 0, stream>>>(ym, qkv_y_w, qkv_y_b, qkvy, 256,  9216, 1536, 0);
    // 4. q/k rmsnorm + rope + pack bf16 head-major
    qkprep_kernel<<<6912, 256, 0, stream>>>(qkvx, qkvy, q_norm_x_w, k_norm_x_w,
                                            q_norm_y_w, k_norm_y_w, rope_cos, rope_sin,
                                            qb, kb, vb);
    // 5. flash attention (WMMA)
    attn_kernel<<<dim3(24, 18), 256, 0, stream>>>(qb, kb, vb, attnb);
    // 6. output projections
    gemm_bf16<<<dim3(16, 24), 256, 0, stream>>>(attnb, proj_x_w, proj_x_b, xattn, 2048, 3072, 3072, 0);
    gemm_bf16<<<dim3(2,  12), 256, 0, stream>>>(attnb + (size_t)2048 * 3072, proj_y_w, proj_y_b,
                                                yattn, 256, 1536, 3072, 0);
    // 7. attention residuals (written into d_out)
    residual_kernel<<<2048, 256, 0, stream>>>(x, xattn, modx + 3072, outx, 3072);
    residual_kernel<<<256,  256, 0, stream>>>(y, yattn, mody + 1536, outy, 1536);
    // 8. MLP x
    modnorm_kernel<<<2048, 256, 0, stream>>>(outx, modx + 2 * 3072, xm, 3072);
    gemm_bf16<<<dim3(16, 128), 256, 0, stream>>>(xm, mlp_x_w1, nullptr, hgx, 2048, 16384, 3072, 1);
    silugate_kernel<<<4096, 256, 0, stream>>>(hgx, actx, 2048, 8192);
    gemm_bf16<<<dim3(16, 24), 256, 0, stream>>>(actx, mlp_x_w2, nullptr, mlpx, 2048, 3072, 8192, 0);
    residual_kernel<<<2048, 256, 0, stream>>>(outx, mlpx, modx + 3 * 3072, outx, 3072);
    // 9. MLP y
    modnorm_kernel<<<256, 256, 0, stream>>>(outy, mody + 2 * 1536, ym, 1536);
    gemm_bf16<<<dim3(2, 64), 256, 0, stream>>>(ym, mlp_y_w1, nullptr, hgy, 256, 8192, 1536, 1);
    silugate_kernel<<<1024, 256, 0, stream>>>(hgy, acty, 256, 4096);
    gemm_bf16<<<dim3(2, 12), 256, 0, stream>>>(acty, mlp_y_w2, nullptr, mlpy, 256, 1536, 4096, 0);
    residual_kernel<<<256, 256, 0, stream>>>(outy, mlpy, mody + 3 * 1536, outy, 1536);
    (void)in_sizes; (void)n_in; (void)out_size; (void)ws_size;
}
